// MaskedGraphAF_9079560864622
// MI455X (gfx1250) — compile-verified, matrix operand order checked by reading, not run on hardware
//
#include <hip/hip_runtime.h>
#include <hip/hip_bf16.h>
#include <math.h>

// Problem constants (from reference)
#define BB   8
#define NN   38
#define NTT  9
#define ETT  4
#define RR   416
#define EE   378
#define HH   128

typedef __attribute__((ext_vector_type(16))) _Float16 v16h;
typedef __attribute__((ext_vector_type(8)))  float    v8f;

union Frag16 { v16h h; unsigned int u[8]; };

// Load one 16x32 f16 WMMA fragment (A from row-major [M][K], or B from
// transposed-weight storage [N][K]); ldh = row stride in halves (must be even).
// Per CDNA5 ISA 16-bit A layout: lanes 0-15 hold K {k0..k0+7, k0+16..k0+23},
// lanes 16-31 hold K {k0+8..k0+15, k0+24..k0+31}, packed 2 halves / dword.
__device__ __forceinline__ v16h load_frag(const _Float16* __restrict__ base,
                                          int ldh, int row0, int k0) {
  const int lane = threadIdx.x & 31;
  const int r = row0 + (lane & 15);
  const unsigned int* p = (const unsigned int*)(base + (size_t)r * ldh);
  const int db = (k0 >> 1) + ((lane & 16) ? 4 : 0);
  Frag16 f;
#pragma unroll
  for (int i = 0; i < 4; ++i) { f.u[i] = p[db + i]; f.u[4 + i] = p[db + 8 + i]; }
  return f.h;
}

__device__ __forceinline__ v8f wmma16(v16h a, v16h b, v8f c) {
  return __builtin_amdgcn_wmma_f32_16x16x32_f16(false, a, false, b,
                                                (short)0, c, false, false);
}

// Convert fp32 [batch][K][Nreal] -> f16 transposed+padded [batch][Npad][K]
__global__ void convert_wT_kernel(const float* __restrict__ in,
                                  _Float16* __restrict__ out,
                                  int batch, int K, int Nreal, int Npad) {
  int total = batch * Npad * K;
  for (int idx = blockIdx.x * blockDim.x + threadIdx.x; idx < total;
       idx += gridDim.x * blockDim.x) {
    int bt = idx / (Npad * K);
    int rem = idx - bt * (Npad * K);
    int n = rem / K;
    int k = rem - n * K;
    float v = (n < Nreal) ? in[(size_t)bt * K * Nreal + (size_t)k * Nreal + n] : 0.f;
    out[idx] = (_Float16)v;
  }
}

// One block per (b, r) graph. 8 waves. Fully-fused 3-layer GCN in LDS,
// emits g_node rows and gathered g_ne rows (pairs + graph emb) as f16.
__global__ __launch_bounds__(256) void gcn_kernel(
    const float* __restrict__ x, const float* __restrict__ adj,
    const unsigned char* __restrict__ mask_node,
    const unsigned char* __restrict__ mask_edge,
    const int* __restrict__ idxsel, const float* __restrict__ emb_w,
    const _Float16* __restrict__ gcWt, const float* __restrict__ gc_b,
    _Float16* __restrict__ gnode, _Float16* __restrict__ gne)
{
  __shared__ _Float16 sh_h[48 * 136];     // h, rows padded 38->48
  __shared__ _Float16 sh_supT[128 * 72];  // sup transposed: [o][m], m padded ->64(+8)
  __shared__ _Float16 sh_am[3 * 48 * 72]; // masked adj per edge type, padded 48x64(+8)

  const int tid  = threadIdx.x;
  const int lane = tid & 31;
  const int w    = tid >> 5;              // wave id 0..7 == output col strip
  const int bid  = blockIdx.x;
  const int b    = bid / RR;
  const int r    = bid % RR;

  // h0 = (masked x) @ emb_w  (K=9: scalar FMA, too small for WMMA)
  for (int idx = tid; idx < 48 * 128; idx += 256) {
    int n = idx >> 7, d = idx & 127;
    float v = 0.f;
    if (n < NN && mask_node[r * NN + n]) {
      const float* xr = x + (size_t)(b * NN + n) * NTT;
#pragma unroll
      for (int t = 0; t < NTT; ++t) v += xr[t] * emb_w[t * HH + d];
    }
    sh_h[n * 136 + d] = (_Float16)v;
  }
  // zero supT pad columns m=48..71 (read as B with K=64; am cols there are 0)
  for (int idx = tid; idx < 128 * 24; idx += 256) {
    int o = idx / 24, m = 48 + idx % 24;
    sh_supT[o * 72 + m] = (_Float16)0.f;
  }
  // masked adjacency (first 3 edge types), zero-padded
  for (int idx = tid; idx < 3 * 48 * 72; idx += 256) {
    int e = idx / (48 * 72);
    int rem = idx - e * (48 * 72);
    int n = rem / 72, m = rem % 72;
    float v = 0.f;
    if (n < NN && m < NN && mask_edge[(size_t)r * (NN * NN) + n * NN + m])
      v = adj[(((size_t)b * ETT + e) * NN + n) * NN + m];
    sh_am[idx] = (_Float16)v;
  }
  __syncthreads();

  for (int l = 0; l < 3; ++l) {
    v8f acc[3] = {};                      // out tile accumulators (3 M-tiles, strip nt=w)
#pragma unroll 1
    for (int e = 0; e < 3; ++e) {
      // sup_e^T = (h @ W_e)^T : wave w produces output cols [w*16, w*16+16)
      const _Float16* Wt = gcWt + (size_t)(l * 3 + e) * HH * HH;
#pragma unroll
      for (int mt = 0; mt < 3; ++mt) {
        v8f c = {};
#pragma unroll
        for (int kt = 0; kt < 4; ++kt) {
          v16h a  = load_frag(sh_h, 136, mt * 16, kt * 32);
          v16h bf = load_frag(Wt, HH, w * 16, kt * 32);
          c = wmma16(a, bf, c);
        }
        const int o = w * 16 + (lane & 15);
        const int mbase = mt * 16 + ((lane & 16) ? 8 : 0);
#pragma unroll
        for (int j = 0; j < 8; ++j)       // transposed store: contiguous in m
          sh_supT[o * 72 + mbase + j] = (_Float16)c[j];
      }
      __syncthreads();
      // acc += am_e @ sup_e   (A = am 48x64 row-major, B = supT as [N][K])
#pragma unroll
      for (int mt = 0; mt < 3; ++mt) {
#pragma unroll
        for (int kt = 0; kt < 2; ++kt) {
          v16h a  = load_frag(sh_am + e * 48 * 72, 72, mt * 16, kt * 32);
          v16h bf = load_frag(sh_supT, 72, w * 16, kt * 32);
          acc[mt] = wmma16(a, bf, acc[mt]);
        }
      }
      __syncthreads();
    }
    // bias + relu (except last layer), write back into h (real rows only)
    const int o = w * 16 + (lane & 15);
    const float bias = gc_b[l * HH + o];
#pragma unroll
    for (int mt = 0; mt < 3; ++mt) {
      const int mbase = mt * 16 + ((lane & 16) ? 8 : 0);
#pragma unroll
      for (int j = 0; j < 8; ++j) {
        int n = mbase + j;
        float v = acc[mt][j] + bias;
        if (l < 2) v = fmaxf(v, 0.f);
        sh_h[n * 136 + o] = (_Float16)((n < NN) ? v : 0.f);
      }
    }
    __syncthreads();
  }

  // graph embedding (sum over 38 nodes) + node-pair gather
  if (tid < HH) {
    const int o = tid;
    float ge = 0.f;
    for (int n = 0; n < NN; ++n) ge += (float)sh_h[n * 136 + o];
    if (r < NN) {
      gnode[(size_t)(b * NN + r) * HH + o] = (_Float16)ge;
    } else {
      const int j = r - NN;
      const int i0 = idxsel[2 * j], i1 = idxsel[2 * j + 1];
      _Float16* dst = gne + (size_t)(b * EE + j) * 384;
      dst[o]       = sh_h[i0 * 136 + o];
      dst[128 + o] = sh_h[i1 * 136 + o];
      dst[256 + o] = (_Float16)ge;
    }
  }
}

// Node coupling: one wave per 16-row tile of g_node (304 rows = 19 tiles).
// A fragments loaded once, reused across all 12 layers.
__global__ __launch_bounds__(32) void node_flow_kernel(
    const _Float16* __restrict__ gnode, const float* __restrict__ x_deq,
    const _Float16* __restrict__ nw1t, const float* __restrict__ nb1,
    const _Float16* __restrict__ nw2t, const float* __restrict__ nb2,
    float* __restrict__ out)
{
  __shared__ _Float16 sh_t[16 * 136];
  __shared__ float    sh_st[16 * 32];
  const int lane = threadIdx.x & 31;
  const int row0 = blockIdx.x * 16;

  v16h afr[4];
#pragma unroll
  for (int kt = 0; kt < 4; ++kt) afr[kt] = load_frag(gnode, HH, row0, kt * 32);

  float xd[NTT] = {};
  float xlj = 0.f;
  const int row = row0 + lane;
  if (lane < 16) {
#pragma unroll
    for (int t = 0; t < NTT; ++t) xd[t] = x_deq[(size_t)row * NTT + t];
  }

  for (int i = 0; i < 12; ++i) {
    // hidden = tanh(A @ nw1 + b1)
#pragma unroll 1
    for (int nt = 0; nt < 8; ++nt) {
      v8f c = {};
#pragma unroll
      for (int kt = 0; kt < 4; ++kt) {
        v16h bf = load_frag(nw1t + (size_t)i * HH * HH, HH, nt * 16, kt * 32);
        c = wmma16(afr[kt], bf, c);
      }
      const int col = nt * 16 + (lane & 15);
      const float b1 = nb1[i * HH + col];
      const int mbase = (lane & 16) ? 8 : 0;
#pragma unroll
      for (int j = 0; j < 8; ++j)
        sh_t[(mbase + j) * 136 + col] = (_Float16)tanhf(c[j] + b1);
    }
    __syncthreads();
    v16h a2[4];
#pragma unroll
    for (int kt = 0; kt < 4; ++kt) a2[kt] = load_frag(sh_t, 136, 0, kt * 32);
#pragma unroll 1
    for (int nt = 0; nt < 2; ++nt) {       // st = hidden @ nw2 + b2 (18 cols, pad 32)
      v8f c = {};
#pragma unroll
      for (int kt = 0; kt < 4; ++kt) {
        v16h bf = load_frag(nw2t + (size_t)i * 32 * HH, HH, nt * 16, kt * 32);
        c = wmma16(a2[kt], bf, c);
      }
      const int col = nt * 16 + (lane & 15);
      const float b2 = (col < 2 * NTT) ? nb2[i * 2 * NTT + col] : 0.f;
      const int mbase = (lane & 16) ? 8 : 0;
#pragma unroll
      for (int j = 0; j < 8; ++j)
        sh_st[(mbase + j) * 32 + col] = c[j] + b2;
    }
    __syncthreads();
    if (lane < 16) {
#pragma unroll
      for (int t = 0; t < NTT; ++t) {
        float s  = 1.f / (1.f + expf(-(sh_st[lane * 32 + t] + 2.f)));
        float tt = sh_st[lane * 32 + NTT + t];
        xd[t] = xd[t] * s + tt;
        xlj += logf(fabsf(s) + 1e-20f);
      }
    }
    __syncthreads();
  }
  if (lane < 16) {
#pragma unroll
    for (int t = 0; t < NTT; ++t) out[(size_t)row * NTT + t] = xd[t];
    atomicAdd(&out[14832 + row / NN], xlj);
  }
}

// Edge coupling: one wave per 16-row tile of g_ne (3024 rows = 189 tiles).
__global__ __launch_bounds__(32) void edge_flow_kernel(
    const _Float16* __restrict__ gne, const float* __restrict__ adj_deq,
    const _Float16* __restrict__ ew1t, const float* __restrict__ eb1,
    const _Float16* __restrict__ ew2t, const float* __restrict__ eb2,
    float* __restrict__ out)
{
  __shared__ _Float16 sh_t[16 * 136];
  __shared__ float    sh_st[16 * 16];
  const int lane = threadIdx.x & 31;
  const int row0 = blockIdx.x * 16;

  v16h afr[12];                            // A = g_ne tile, 16x384, reused 12 layers
#pragma unroll
  for (int kt = 0; kt < 12; ++kt) afr[kt] = load_frag(gne, 384, row0, kt * 32);

  float ad[ETT] = {};
  float alj = 0.f;
  const int row = row0 + lane;
  if (lane < 16) {
#pragma unroll
    for (int t = 0; t < ETT; ++t) ad[t] = adj_deq[(size_t)row * ETT + t];
  }

  for (int i = 0; i < 12; ++i) {
#pragma unroll 1
    for (int nt = 0; nt < 8; ++nt) {       // hidden = tanh(A @ ew1 + b1)
      v8f c = {};
#pragma unroll
      for (int kt = 0; kt < 12; ++kt) {
        v16h bf = load_frag(ew1t + (size_t)i * HH * 384, 384, nt * 16, kt * 32);
        c = wmma16(afr[kt], bf, c);
      }
      const int col = nt * 16 + (lane & 15);
      const float b1 = eb1[i * HH + col];
      const int mbase = (lane & 16) ? 8 : 0;
#pragma unroll
      for (int j = 0; j < 8; ++j)
        sh_t[(mbase + j) * 136 + col] = (_Float16)tanhf(c[j] + b1);
    }
    __syncthreads();
    {
      v16h a2[4];
#pragma unroll
      for (int kt = 0; kt < 4; ++kt) a2[kt] = load_frag(sh_t, 136, 0, kt * 32);
      v8f c = {};                          // st: only 8 real cols -> one N-tile
#pragma unroll
      for (int kt = 0; kt < 4; ++kt) {
        v16h bf = load_frag(ew2t + (size_t)i * 32 * HH, HH, 0, kt * 32);
        c = wmma16(a2[kt], bf, c);
      }
      const int col = lane & 15;
      const float b2 = (col < 2 * ETT) ? eb2[i * 2 * ETT + col] : 0.f;
      const int mbase = (lane & 16) ? 8 : 0;
#pragma unroll
      for (int j = 0; j < 8; ++j)
        sh_st[(mbase + j) * 16 + col] = c[j] + b2;
    }
    __syncthreads();
    if (lane < 16) {
#pragma unroll
      for (int t = 0; t < ETT; ++t) {
        float s  = 1.f / (1.f + expf(-(sh_st[lane * 16 + t] + 2.f)));
        float tt = sh_st[lane * 16 + ETT + t];
        ad[t] = ad[t] * s + tt;
        alj += logf(fabsf(s) + 1e-20f);
      }
    }
    __syncthreads();
  }
  if (lane < 16) {
#pragma unroll
    for (int t = 0; t < ETT; ++t) out[2736 + (size_t)row * ETT + t] = ad[t];
    atomicAdd(&out[14840 + row / EE], alj);
  }
}

extern "C" void kernel_launch(void* const* d_in, const int* in_sizes, int n_in,
                              void* d_out, int out_size, void* d_ws, size_t ws_size,
                              hipStream_t stream) {
  (void)in_sizes; (void)n_in; (void)out_size; (void)ws_size;
  const float* x        = (const float*)d_in[0];
  const float* adj      = (const float*)d_in[1];
  const float* x_deq    = (const float*)d_in[2];
  const float* adj_deq  = (const float*)d_in[3];
  const unsigned char* mask_node = (const unsigned char*)d_in[4];
  const unsigned char* mask_edge = (const unsigned char*)d_in[5];
  const int*   idxsel   = (const int*)d_in[6];
  const float* emb_w    = (const float*)d_in[7];
  const float* gc_w     = (const float*)d_in[8];
  const float* gc_b     = (const float*)d_in[9];
  const float* nw1      = (const float*)d_in[10];
  const float* nb1      = (const float*)d_in[11];
  const float* nw2      = (const float*)d_in[12];
  const float* nb2      = (const float*)d_in[13];
  const float* ew1      = (const float*)d_in[14];
  const float* eb1      = (const float*)d_in[15];
  const float* ew2      = (const float*)d_in[16];
  const float* eb2      = (const float*)d_in[17];
  float* out = (float*)d_out;

  // workspace carve-out (~4.5 MB total)
  char* ws = (char*)d_ws;
  size_t off = 0;
  auto carve = [&](size_t bytes) {
    char* p = ws + off;
    off = (off + bytes + 255) & ~(size_t)255;
    return p;
  };
  _Float16* gcWt  = (_Float16*)carve((size_t)9  * 128 * 128 * 2);
  _Float16* nw1t  = (_Float16*)carve((size_t)12 * 128 * 128 * 2);
  _Float16* nw2t  = (_Float16*)carve((size_t)12 * 32  * 128 * 2);
  _Float16* ew1t  = (_Float16*)carve((size_t)12 * 128 * 384 * 2);
  _Float16* ew2t  = (_Float16*)carve((size_t)12 * 32  * 128 * 2);
  _Float16* gnode = (_Float16*)carve((size_t)304  * 128 * 2);
  _Float16* gne   = (_Float16*)carve((size_t)3024 * 384 * 2);

  // zero the xlj/alj accumulation slots (d_out is poisoned before timing)
  hipMemsetAsync(out + 14832, 0, 16 * sizeof(float), stream);

  // fp32 -> transposed/padded f16 weight conversion
  convert_wT_kernel<<<144, 256, 0, stream>>>(gc_w, gcWt, 9, 128, 128, 128);
  convert_wT_kernel<<<192, 256, 0, stream>>>(nw1, nw1t, 12, 128, 128, 128);
  convert_wT_kernel<<<48,  256, 0, stream>>>(nw2, nw2t, 12, 128, 18, 32);
  convert_wT_kernel<<<576, 256, 0, stream>>>(ew1, ew1t, 12, 384, 128, 128);
  convert_wT_kernel<<<48,  256, 0, stream>>>(ew2, ew2t, 12, 128, 8, 32);

  // fused per-graph GCN + embedding gather
  gcn_kernel<<<BB * RR, 256, 0, stream>>>(x, adj, mask_node, mask_edge, idxsel,
                                          emb_w, gcWt, gc_b, gnode, gne);
  // coupling flows
  node_flow_kernel<<<19,  32, 0, stream>>>(gnode, x_deq, nw1t, nb1, nw2t, nb2, out);
  edge_flow_kernel<<<189, 32, 0, stream>>>(gne, adj_deq, ew1t, eb1, ew2t, eb2, out);
}
